// ResidualBlock_39616778338828
// MI455X (gfx1250) — matrix operand order repaired and checked by reading, not compiled, not run on hardware
//
#include <hip/hip_runtime.h>
#include <hip/hip_bf16.h>

// ---------------------------------------------------------------------------
// Fused ResidualBlock for MI455X (gfx1250, wave32, WMMA):
//   out = LReLU(LN(x)) @ We^T + be  -> LReLU(LN(.)) @ Ws^T + bs  -> + x
// 32 rows per block, 256 threads (8 wave32). ~87 KB LDS -> 3 blocks/WGP
// (24 waves) for latency hiding. Matmuls: v_wmma_f32_16x16x32_bf16 with
// 8 (GEMM1) / 4 (GEMM2) independent accumulator tiles per A-fragment.
// ---------------------------------------------------------------------------

#define N_NODES 262144
#define D_IN    256
#define D_EXP   512
#define D_OUT   256
#define M_TILE  32
#define LN_EPS  1e-5f
#define SLOPE   0.01f

typedef __attribute__((ext_vector_type(16))) __bf16 v16bf;
typedef __attribute__((ext_vector_type(8)))  float  v8f;

// LDS layout (bytes). Rows padded to break bank-conflict strides.
#define A1_STRIDE 272   // u16/row (256 + 16 pad)  -> 544 B
#define H1_STRIDE 528   // u16/row (512 + 16 pad)  -> 1056 B
#define X_STRIDE  260   // f32/row (256 + 4 pad)   -> 1040 B
#define OFF_A1    0
#define OFF_H1    (OFF_A1 + M_TILE * A1_STRIDE * 2)     // 17408
#define OFF_X     (OFF_H1 + M_TILE * H1_STRIDE * 2)     // 51200
#define OFF_STATS (OFF_X  + M_TILE * X_STRIDE  * 4)     // 84480 (mean1,rstd1,mean2,rstd2)
#define OFF_PART  (OFF_STATS + 4 * M_TILE * 4)          // 84992 (256 x {sum,sumsq})
#define SMEM_BYTES (OFF_PART + 256 * 2 * 4)             // 87040 B -> 3 blocks / 320 KB WGP

__device__ __forceinline__ unsigned short f2bf(float f) {
    unsigned int u = __float_as_uint(f);
    u += 0x7FFFu + ((u >> 16) & 1u);       // round-to-nearest-even
    return (unsigned short)(u >> 16);
}
__device__ __forceinline__ unsigned int pack2bf(float lo, float hi) {
    return (unsigned int)f2bf(lo) | ((unsigned int)f2bf(hi) << 16);
}
__device__ __forceinline__ float bf_lo(unsigned int u) { return __uint_as_float(u << 16); }
__device__ __forceinline__ float bf_hi(unsigned int u) { return __uint_as_float(u & 0xFFFF0000u); }
__device__ __forceinline__ float lrelu(float v) { return v > 0.0f ? v : SLOPE * v; }

union FragBF { uint4 q[2]; v16bf v; };   // 16 bf16 = 32 B = one wmma A/B operand

__global__ __launch_bounds__(256)
void resblock_fused_kernel(const float* __restrict__ x,
                           const float* __restrict__ w1, const float* __restrict__ b1,
                           const float* __restrict__ be,
                           const float* __restrict__ w2, const float* __restrict__ b2,
                           const float* __restrict__ bs,
                           const unsigned short* __restrict__ WeB,  // [512,256] bf16, row-major
                           const unsigned short* __restrict__ WsB,  // [256,512] bf16, row-major
                           float* __restrict__ out)
{
    extern __shared__ char smem[];
    unsigned short* A1 = (unsigned short*)(smem + OFF_A1);  // LN1+LReLU(x), bf16
    unsigned short* H1 = (unsigned short*)(smem + OFF_H1);  // GEMM1 out (then LN2'd), bf16
    float* X     = (float*)(smem + OFF_X);                  // raw x tile (residual)
    float* mean1 = (float*)(smem + OFF_STATS);
    float* rstd1 = mean1 + M_TILE;
    float* mean2 = mean1 + 2 * M_TILE;
    float* rstd2 = mean1 + 3 * M_TILE;
    float* part  = (float*)(smem + OFF_PART);

    const int tid   = threadIdx.x;
    const int lane  = tid & 31;
    const int wave  = tid >> 5;
    const int mSub  = wave & 1;    // which 16-row sub-tile this wave computes
    const int nQ    = wave >> 1;   // which quarter of the N dimension
    const int laneN = lane & 15;   // column within a 16x16 tile
    const int kSel  = lane >> 4;   // ISA lane-half selector for A/B/C layouts
    const long rows0 = (long)blockIdx.x * M_TILE;

    // ---- Phase 1a: stream x tile -> LDS (coalesced; kept for residual) -----
    #pragma unroll
    for (int it = 0; it < 8; ++it) {
        int flat = it * 1024 + tid * 4;          // 32*256 floats, float4 chunks
        int r = flat >> 8, c = flat & 255;
        float4 v = *(const float4*)(x + rows0 * D_IN + flat);
        *(float4*)(X + r * X_STRIDE + c) = v;
    }
    __syncthreads();

    // ---- Phase 1b: LN1 row stats (8 threads / row, float4) -----------------
    {
        int r = tid >> 3, cb = (tid & 7) * 32;
        const float* row = X + r * X_STRIDE + cb;
        float s = 0.f, ss = 0.f;
        #pragma unroll
        for (int j = 0; j < 8; ++j) {
            float4 v = *(const float4*)(row + j * 4);
            s  += v.x + v.y + v.z + v.w;
            ss += v.x * v.x + v.y * v.y + v.z * v.z + v.w * v.w;
        }
        part[tid * 2] = s; part[tid * 2 + 1] = ss;
    }
    __syncthreads();
    if (tid < M_TILE) {
        float s = 0.f, ss = 0.f;
        #pragma unroll
        for (int q = 0; q < 8; ++q) { s += part[(tid*8+q)*2]; ss += part[(tid*8+q)*2+1]; }
        float m   = s * (1.0f / D_IN);
        float var = ss * (1.0f / D_IN) - m * m;
        mean1[tid] = m; rstd1[tid] = rsqrtf(var + LN_EPS);
    }
    __syncthreads();

    // ---- Phase 1c: A1 = bf16(LReLU(LN1(x))), packed pair writes ------------
    {
        int r = tid >> 3, cb = (tid & 7) * 32;
        float m = mean1[r], rs = rstd1[r];
        const float* row = X + r * X_STRIDE + cb;
        unsigned int* arow = (unsigned int*)(A1 + r * A1_STRIDE + cb);
        #pragma unroll
        for (int j = 0; j < 8; ++j) {
            float4 xv = *(const float4*)(row + j * 4);
            float4 wv = *(const float4*)(w1 + cb + j * 4);
            float4 bv = *(const float4*)(b1 + cb + j * 4);
            float v0 = lrelu((xv.x - m) * rs * wv.x + bv.x);
            float v1 = lrelu((xv.y - m) * rs * wv.y + bv.y);
            float v2 = lrelu((xv.z - m) * rs * wv.z + bv.z);
            float v3 = lrelu((xv.w - m) * rs * wv.w + bv.w);
            arow[j * 2]     = pack2bf(v0, v1);
            arow[j * 2 + 1] = pack2bf(v2, v3);
        }
    }
    __syncthreads();

    // ---- Phase 2: GEMM1  h1[32,512] = A1 @ We^T + be  (bf16 WMMA) ----------
    // A frag (16x32 bf16): lane row = mSub*16 + lane%16; K chunks split by lane half.
    // B frag (32x16 bf16): lane col = We row; K chunks split by lane half (same as A).
    {
        const int mRow = mSub * 16 + laneN;
        v8f acc[8];
        #pragma unroll
        for (int t = 0; t < 8; ++t) acc[t] = (v8f){};
        #pragma unroll 1
        for (int k8 = 0; k8 < 8; ++k8) {              // K = 256 in steps of 32
            int kBase = k8 * 32;
            FragBF a;
            a.q[0] = *(const uint4*)(A1 + mRow * A1_STRIDE + kBase + kSel * 8);
            a.q[1] = *(const uint4*)(A1 + mRow * A1_STRIDE + kBase + 16 + kSel * 8);
            #pragma unroll
            for (int t = 0; t < 8; ++t) {             // 8 independent acc tiles
                int col = (nQ * 8 + t) * 16 + laneN;
                FragBF b;
                const uint4* bp = (const uint4*)(WeB + col * D_IN + kBase + kSel * 8);
                b.q[0] = bp[0]; b.q[1] = bp[2];
                acc[t] = __builtin_amdgcn_wmma_f32_16x16x32_bf16(
                             false, a.v, false, b.v, (short)0, acc[t], false, false);
            }
        }
        #pragma unroll
        for (int t = 0; t < 8; ++t) {                 // epilogue: +be, bf16 -> H1
            int col = (nQ * 8 + t) * 16 + laneN;
            float bev = be[col];
            #pragma unroll
            for (int i = 0; i < 8; ++i) {
                int m = mSub * 16 + kSel * 8 + i;     // C layout: lane half picks M 0-7/8-15
                H1[m * H1_STRIDE + col] = f2bf(acc[t][i] + bev);
            }
        }
    }
    __syncthreads();

    // ---- Phase 3: LN2 stats over H1, then LN2+LReLU in place (packed) ------
    {
        int r = tid >> 3, cb = (tid & 7) * 64;
        const unsigned int* row = (const unsigned int*)(H1 + r * H1_STRIDE + cb);
        float s = 0.f, ss = 0.f;
        #pragma unroll
        for (int j = 0; j < 16; ++j) {                // 64 bf16 as 16x uint2
            uint2 u = *(const uint2*)(row + j * 2);
            float v0 = bf_lo(u.x), v1 = bf_hi(u.x), v2 = bf_lo(u.y), v3 = bf_hi(u.y);
            s  += v0 + v1 + v2 + v3;
            ss += v0 * v0 + v1 * v1 + v2 * v2 + v3 * v3;
        }
        part[tid * 2] = s; part[tid * 2 + 1] = ss;
    }
    __syncthreads();
    if (tid < M_TILE) {
        float s = 0.f, ss = 0.f;
        #pragma unroll
        for (int q = 0; q < 8; ++q) { s += part[(tid*8+q)*2]; ss += part[(tid*8+q)*2+1]; }
        float m   = s * (1.0f / D_EXP);
        float var = ss * (1.0f / D_EXP) - m * m;
        mean2[tid] = m; rstd2[tid] = rsqrtf(var + LN_EPS);
    }
    __syncthreads();
    {
        int r = tid >> 3, cb = (tid & 7) * 64;
        float m = mean2[r], rs = rstd2[r];
        unsigned int* row = (unsigned int*)(H1 + r * H1_STRIDE + cb);
        #pragma unroll
        for (int j = 0; j < 16; ++j) {
            uint2 u = *(const uint2*)(row + j * 2);
            float4 wv = *(const float4*)(w2 + cb + j * 4);
            float4 bv = *(const float4*)(b2 + cb + j * 4);
            float v0 = lrelu((bf_lo(u.x) - m) * rs * wv.x + bv.x);
            float v1 = lrelu((bf_hi(u.x) - m) * rs * wv.y + bv.y);
            float v2 = lrelu((bf_lo(u.y) - m) * rs * wv.z + bv.z);
            float v3 = lrelu((bf_hi(u.y) - m) * rs * wv.w + bv.w);
            uint2 o; o.x = pack2bf(v0, v1); o.y = pack2bf(v2, v3);
            *(uint2*)(row + j * 2) = o;
        }
    }
    __syncthreads();

    // ---- Phase 4: GEMM2  out[32,256] = H1 @ Ws^T + bs + x ------------------
    {
        const int mRow = mSub * 16 + laneN;
        v8f acc[4];
        #pragma unroll
        for (int t = 0; t < 4; ++t) acc[t] = (v8f){};
        #pragma unroll 1
        for (int k8 = 0; k8 < 16; ++k8) {             // K = 512 in steps of 32
            int kBase = k8 * 32;
            FragBF a;
            a.q[0] = *(const uint4*)(H1 + mRow * H1_STRIDE + kBase + kSel * 8);
            a.q[1] = *(const uint4*)(H1 + mRow * H1_STRIDE + kBase + 16 + kSel * 8);
            #pragma unroll
            for (int t = 0; t < 4; ++t) {
                int col = (nQ * 4 + t) * 16 + laneN;
                FragBF b;
                const uint4* bp = (const uint4*)(WsB + col * D_EXP + kBase + kSel * 8);
                b.q[0] = bp[0]; b.q[1] = bp[2];
                acc[t] = __builtin_amdgcn_wmma_f32_16x16x32_bf16(
                             false, a.v, false, b.v, (short)0, acc[t], false, false);
            }
        }
        #pragma unroll
        for (int t = 0; t < 4; ++t) {                 // epilogue: +bs, +identity, store
            int col = (nQ * 4 + t) * 16 + laneN;
            float bsv = bs[col];
            #pragma unroll
            for (int i = 0; i < 8; ++i) {
                int m = mSub * 16 + kSel * 8 + i;
                out[(rows0 + m) * D_OUT + col] = acc[t][i] + bsv + X[m * X_STRIDE + col];
            }
        }
    }
}

// Prologue: convert f32 weights to bf16 copies in workspace (512 KB total,
// L2-resident for the whole run at 192 MB L2).
__global__ __launch_bounds__(256)
void convert_weights_kernel(const float* __restrict__ We, const float* __restrict__ Ws,
                            unsigned short* __restrict__ WeB, unsigned short* __restrict__ WsB)
{
    int i = blockIdx.x * 256 + threadIdx.x;   // 131072 elements each
    WeB[i] = f2bf(We[i]);
    WsB[i] = f2bf(Ws[i]);
}

extern "C" void kernel_launch(void* const* d_in, const int* in_sizes, int n_in,
                              void* d_out, int out_size, void* d_ws, size_t ws_size,
                              hipStream_t stream)
{
    const float* x  = (const float*)d_in[0];
    // d_in[1] = node_separation: does not affect the math (per-graph loop is a no-op)
    const float* w1 = (const float*)d_in[2];
    const float* b1 = (const float*)d_in[3];
    const float* We = (const float*)d_in[4];
    const float* be = (const float*)d_in[5];
    const float* w2 = (const float*)d_in[6];
    const float* b2 = (const float*)d_in[7];
    const float* Ws = (const float*)d_in[8];
    const float* bs = (const float*)d_in[9];
    float* out = (float*)d_out;

    unsigned short* WeB = (unsigned short*)d_ws;           // 512*256 bf16
    unsigned short* WsB = WeB + D_EXP * D_IN;              // 256*512 bf16 (offset 256 KB)

    hipLaunchKernelGGL(convert_weights_kernel, dim3(512), dim3(256), 0, stream,
                       We, Ws, WeB, WsB);
    hipLaunchKernelGGL(resblock_fused_kernel, dim3(N_NODES / M_TILE), dim3(256),
                       SMEM_BYTES, stream,
                       x, w1, b1, be, w2, b2, bs, WeB, WsB, out);
}
